// LMI_85452669321593
// MI455X (gfx1250) — matrix operand order repaired and checked by reading, not compiled, not run on hardware
//
#include <hip/hip_runtime.h>
#include <hip/hip_bf16.h>

// ---------------------------------------------------------------------------
// LIIF-style forward on MI455X (gfx1250, wave32, WMMA).
// Heavy GEMMs (h@w_m2 [64x1024], feats@w_i1 [70x64]) run on
// v_wmma_f32_16x16x32_f16 with f32 accumulation.
// ---------------------------------------------------------------------------

#define BB   2
#define QQ   32768
#define CC   64
#define HH_  96
#define WW_  96
#define HID  64
#define KK   16

typedef _Float16 half_t;
typedef __attribute__((ext_vector_type(4)))  _Float16 v4h;
typedef __attribute__((ext_vector_type(8)))  _Float16 v8h;
typedef __attribute__((ext_vector_type(16))) _Float16 v16h;
typedef __attribute__((ext_vector_type(8)))  float    v8f;

#define SHUF16(lo, hi) __builtin_shufflevector(lo, hi, 0,1,2,3,4,5,6,7,8,9,10,11,12,13,14,15)

// ---- workspace layout (halves) --------------------------------------------
// feat_hwc : [B][H][W][C] f16                       offset 0
// w_m2T    : [1024][64]  f16 (K-contiguous)         offset FEAT_HALF
// w_i1T    : [64][96]    f16 (K-contiguous, K 70->96 zero padded)
#define FEAT_HALF  (BB * HH_ * WW_ * CC)        // 1,179,648
#define WM2T_OFF   FEAT_HALF
#define WM2T_HALF  (1024 * 64)
#define WI1T_OFF   (WM2T_OFF + WM2T_HALF)
#define WI1T_HALF  (64 * 96)

// ---------------------------------------------------------------------------
// Kernel 1: 3x3 SAME conv (cross-correlation, NCHW in, HWC-f16 out)
// ---------------------------------------------------------------------------
__global__ void liif_conv3x3(const float* __restrict__ inp,
                             const float* __restrict__ w_enc,
                             const float* __restrict__ b_enc,
                             half_t* __restrict__ feat) {
    int idx = blockIdx.x * 256 + threadIdx.x;
    if (idx >= BB * HH_ * WW_ * CC) return;
    int c = idx & 63;
    int t = idx >> 6;
    int x = t % WW_; t /= WW_;
    int y = t % HH_;
    int b = t / HH_;
    float acc = b_enc[c];
    #pragma unroll
    for (int ic = 0; ic < 3; ++ic)
        #pragma unroll
        for (int dy = -1; dy <= 1; ++dy)
            #pragma unroll
            for (int dx = -1; dx <= 1; ++dx) {
                int yy = y + dy, xx = x + dx;
                if (yy >= 0 && yy < HH_ && xx >= 0 && xx < WW_)
                    acc += inp[((b * 3 + ic) * HH_ + yy) * WW_ + xx] *
                           w_enc[((c * 3 + ic) * 3 + (dy + 1)) * 3 + (dx + 1)];
            }
    feat[((size_t)((b * HH_ + y) * WW_ + x)) * CC + c] = (half_t)acc;
}

// ---------------------------------------------------------------------------
// Kernel 2: weight prep -> f16, transposed so K is contiguous per output col.
// ---------------------------------------------------------------------------
__global__ void liif_prep_weights(const float* __restrict__ w_m2,   // [64][1024]
                                  const float* __restrict__ w_i1,   // [70][64]
                                  half_t* __restrict__ w_m2T,       // [1024][64]
                                  half_t* __restrict__ w_i1T) {     // [64][96]
    int idx = blockIdx.x * 256 + threadIdx.x;
    if (idx < 1024 * 64) {
        int n = idx >> 6, k = idx & 63;
        w_m2T[n * 64 + k] = (half_t)w_m2[k * 1024 + n];
    } else {
        int i2 = idx - 1024 * 64;
        if (i2 < 64 * 96) {
            int n = i2 / 96, k = i2 % 96;
            w_i1T[n * 96 + k] = (half_t)((k < 70) ? w_i1[k * 64 + n] : 0.0f);
        }
    }
}

// ---------------------------------------------------------------------------
// Kernel 3: fused meta-MLP + modulated imnet + area blend.
// One workgroup (256 thr = 8 wave32) handles 16 queries of one batch.
// ---------------------------------------------------------------------------
// LDS layout (bytes):
//   sA3   [256][96] f16  : 49152   (imnet A matrix, K padded to 96)
//   sQF   [256][64] f16  : 32768   (gathered q_feat; aliased by sHH in phase 3)
//   sH    [16][64]  f16  :  2048   (meta hidden, f16 for WMMA A)
//   sMeta [16][33]  f32  :  2112
//   sArea [16][16]  f32  :  1024
//   sPred [256][3]  f32  :  3072
#define SM_A3    0
#define SM_QF    49152
#define SM_H     (49152 + 32768)
#define SM_META  (SM_H + 2048)
#define SM_AREA  (SM_META + 2112)
#define SM_PRED  (SM_AREA + 1024)
#define SM_TOTAL (SM_PRED + 3072)     // 90176 B

__global__ void __launch_bounds__(256)
liif_main(const float* __restrict__ inp,
          const float* __restrict__ coord,
          const float* __restrict__ cell,
          const float* __restrict__ w_m1,   // [33][64]
          const float* __restrict__ b_m1,   // [64]
          const float* __restrict__ b_m2,   // [1024]
          const float* __restrict__ b_i1,   // [64]
          const float* __restrict__ w_i2,   // [64][3]
          const float* __restrict__ b_i2,   // [3]
          const half_t* __restrict__ feat,  // [B][H][W][C] f16
          const half_t* __restrict__ w_m2T, // [1024][64] f16
          const half_t* __restrict__ w_i1T, // [64][96]  f16
          float* __restrict__ out) {        // [B][Q][3]
    __shared__ __align__(32) unsigned char smem[SM_TOTAL];
    half_t* sA3   = (half_t*)(smem + SM_A3);
    half_t* sQF   = (half_t*)(smem + SM_QF);
    half_t* sHH   = sQF;                       // alias: qfeat dead after phase2b
    half_t* sH    = (half_t*)(smem + SM_H);
    float*  sMeta = (float*)(smem + SM_META);
    float*  sArea = (float*)(smem + SM_AREA);
    float*  sPred = (float*)(smem + SM_PRED);

    const int j     = threadIdx.x;             // 0..255
    const int wg    = blockIdx.x;              // 0..4095
    const int b     = wg >> 11;                // / (Q/16)
    const int qbase = (wg & 2047) << 4;
    const int ql    = j >> 4;
    const int kk    = j & 15;
    const int q     = qbase + ql;

    // ---------------- phase 0: coords, gathers, A3 tail ----------------
    {
        const float cy    = coord[((size_t)(b * QQ) + q) * 2 + 0];
        const float cx    = coord[((size_t)(b * QQ) + q) * 2 + 1];
        const float rcell = cell[((size_t)(b * QQ) + q) * 2 + 0] * (float)HH_;

        const float svy = 2.0f * (float)(kk >> 2) - 3.0f;   // v = {-3,-1,1,3}, vx outer
        const float svx = 2.0f * (float)(kk & 3)  - 3.0f;   // vy inner
        float csy = cy + svy * (1.0f / HH_) + 1e-6f;
        float csx = cx + svx * (1.0f / WW_) + 1e-6f;
        csy = fminf(fmaxf(csy, -1.0f + 1e-6f), 1.0f - 1e-6f);
        csx = fminf(fmaxf(csx, -1.0f + 1e-6f), 1.0f - 1e-6f);
        int iy = (int)rintf((csy + 1.0f) * (HH_ * 0.5f) - 0.5f);   // RNE == jnp.round
        int ix = (int)rintf((csx + 1.0f) * (WW_ * 0.5f) - 0.5f);
        iy = min(max(iy, 0), HH_ - 1);
        ix = min(max(ix, 0), WW_ - 1);

        const float qcy  = -1.0f + (2.0f * iy + 1.0f) / (float)HH_;
        const float qcx  = -1.0f + (2.0f * ix + 1.0f) / (float)WW_;
        const float rel0 = (cy - qcy) * (float)HH_;
        const float rel1 = (cx - qcx) * (float)WW_;
        const float area = fabsf(rel0 * rel1) + 1e-9f;

        // gather 64 contiguous f16 feature channels (128B)
        const half_t* fr = feat + ((size_t)((b * HH_ + iy) * WW_ + ix)) * CC;
        #pragma unroll
        for (int c0 = 0; c0 < 64; c0 += 8)
            *(v8h*)&sQF[j * 64 + c0] = *(const v8h*)&fr[c0];

        // A3 tail: rgb, rel, rel_cell packed as one v8h; zero pad to K=96
        const size_t pix = (size_t)iy * WW_ + ix;
        v8h tail;
        tail[0] = (half_t)inp[((size_t)(b * 3 + 0) * HH_ * WW_) + pix];
        tail[1] = (half_t)inp[((size_t)(b * 3 + 1) * HH_ * WW_) + pix];
        tail[2] = (half_t)inp[((size_t)(b * 3 + 2) * HH_ * WW_) + pix];
        tail[3] = (half_t)rel0;
        tail[4] = (half_t)rel1;
        tail[5] = (half_t)rcell;
        tail[6] = (half_t)0.0f;
        tail[7] = (half_t)0.0f;
        *(v8h*)&sA3[j * 96 + 64] = tail;
        const v8h zero8 = (v8h)(half_t)0.0f;
        *(v8h*)&sA3[j * 96 + 72] = zero8;
        *(v8h*)&sA3[j * 96 + 80] = zero8;
        *(v8h*)&sA3[j * 96 + 88] = zero8;

        sMeta[ql * 33 + 2 * kk + 0] = rel0;
        sMeta[ql * 33 + 2 * kk + 1] = rel1;
        if (kk == 0) sMeta[ql * 33 + 32] = rcell;
        sArea[ql * 16 + kk] = area;
    }
    __syncthreads();

    // ---------------- phase 1: GEMM1 [16,33]@[33,64] (VALU, tiny) -------
    #pragma unroll
    for (int o = j; o < 16 * 64; o += 256) {
        const int qq = o >> 6, n = o & 63;
        float acc = b_m1[n];
        for (int t = 0; t < 33; ++t)
            acc += sMeta[qq * 33 + t] * w_m1[t * 64 + n];
        sH[qq * 64 + n] = (half_t)fmaxf(acc, 0.0f);
    }
    __syncthreads();

    // ---------------- phase 2: GEMM2 via WMMA -> mod columns of A3 ------
    // meta_mix col n = c*16+k  ->  16-wide N-tile t is channel c=t over all k.
    const int wave = j >> 5, lane = j & 31;
    const int ln   = lane & 15;
    const int hi   = lane >> 4;
    const int kb   = hi * 8;        // unused; A-frag also loads at kb16
    const int kb16 = hi * 16;       // B-frag K base
    {
        v16h a0 = *(const v16h*)&sH[ln * 64 + 0  + kb16]; // K 0..31
        v16h a1 = *(const v16h*)&sH[ln * 64 + 32 + kb16]; // K 32..63

        for (int t = wave * 8; t < wave * 8 + 8; ++t) {
            const int ng = t * 16 + ln;                 // global column (lane = N)
            v16h b0 = *(const v16h*)&w_m2T[ng * 64 + 0  + kb16];
            v16h b1 = *(const v16h*)&w_m2T[ng * 64 + 32 + kb16];
            const float bias = b_m2[ng];
            v8f acc;
            #pragma unroll
            for (int r = 0; r < 8; ++r) acc[r] = bias;
            acc = __builtin_amdgcn_wmma_f32_16x16x32_f16(false, a0, false, b0,
                                                         (short)0, acc, false, false);
            acc = __builtin_amdgcn_wmma_f32_16x16x32_f16(false, a1, false, b1,
                                                         (short)0, acc, false, false);
            // D(m=q, n=k) = mod[q][k][c=t]; store f16 column t (no LDS reads)
            #pragma unroll
            for (int r = 0; r < 8; ++r) {
                const int qrow = r + hi * 8;            // query (D-matrix M)
                const int row  = qrow * 16 + ln;        // (q,k) row of A3
                sA3[row * 96 + t] = (half_t)acc[r];
            }
        }
    }
    __syncthreads();

    // ---------------- phase 2b: vectorized modulate A3[:,0:64] *= qfeat -
    {
        half_t* a3r = &sA3[j * 96];
        const half_t* qfr = &sQF[j * 64];
        #pragma unroll
        for (int c0 = 0; c0 < 64; c0 += 8) {
            v8h m = *(const v8h*)&a3r[c0];
            v8h f = *(const v8h*)&qfr[c0];
            *(v8h*)&a3r[c0] = m * f;                    // v_pk_mul_f16
        }
    }
    __syncthreads();

    // ---------------- phase 3: GEMM3 via WMMA [256,96]@[96,64] ----------
    #pragma unroll
    for (int mi = 0; mi < 2; ++mi) {
        const int mt   = wave + mi * 8;                 // M-tile 0..15
        const int rowb = (mt * 16 + ln) * 96;
        v16h af[3];
        #pragma unroll
        for (int s = 0; s < 3; ++s) {
            const int kc = s * 32;
            af[s] = *(const v16h*)&sA3[rowb + kc + kb16];
        }
        #pragma unroll
        for (int nt = 0; nt < 4; ++nt) {
            const int ng = nt * 16 + ln;
            const float bias = b_i1[ng];
            v8f acc;
            #pragma unroll
            for (int r = 0; r < 8; ++r) acc[r] = bias;
            #pragma unroll
            for (int s = 0; s < 3; ++s) {
                v16h bf = *(const v16h*)&w_i1T[ng * 96 + s * 32 + kb16];
                acc = __builtin_amdgcn_wmma_f32_16x16x32_f16(false, af[s], false, bf,
                                                             (short)0, acc, false, false);
            }
            #pragma unroll
            for (int r = 0; r < 8; ++r) {
                const int row = mt * 16 + r + hi * 8;
                sHH[row * 64 + ng] = (half_t)fmaxf(acc[r], 0.0f);
            }
        }
    }
    __syncthreads();

    // ---------------- phase 4: GEMM4 [256,64]@[64,3] (VALU) -------------
    {
        float acc0 = b_i2[0], acc1 = b_i2[1], acc2 = b_i2[2];
        #pragma unroll
        for (int n0 = 0; n0 < 64; n0 += 8) {
            v8h hv = *(const v8h*)&sHH[j * 64 + n0];    // ds_load_b128
            #pragma unroll
            for (int r = 0; r < 8; ++r) {
                const float hval = (float)hv[r];
                acc0 += hval * w_i2[(n0 + r) * 3 + 0];  // uniform -> s_load
                acc1 += hval * w_i2[(n0 + r) * 3 + 1];
                acc2 += hval * w_i2[(n0 + r) * 3 + 2];
            }
        }
        sPred[j * 3 + 0] = acc0;
        sPred[j * 3 + 1] = acc1;
        sPred[j * 3 + 2] = acc2;
    }
    __syncthreads();

    // ---------------- phase 5: reversed-area blend ----------------------
    if (j < 48) {
        const int qq = j / 3, ch = j % 3;
        float tot = 0.0f;
        #pragma unroll
        for (int t = 0; t < 16; ++t) tot += sArea[qq * 16 + t];
        float acc = 0.0f;
        #pragma unroll
        for (int t = 0; t < 16; ++t)
            acc += sPred[(qq * 16 + t) * 3 + ch] * sArea[qq * 16 + 15 - t];
        out[((size_t)(b * QQ) + qbase + qq) * 3 + ch] = acc / tot;
    }
}

// ---------------------------------------------------------------------------
extern "C" void kernel_launch(void* const* d_in, const int* in_sizes, int n_in,
                              void* d_out, int out_size, void* d_ws, size_t ws_size,
                              hipStream_t stream) {
    (void)in_sizes; (void)n_in; (void)out_size; (void)ws_size;
    const float* inp   = (const float*)d_in[0];
    const float* coord = (const float*)d_in[1];
    const float* cell  = (const float*)d_in[2];
    const float* w_enc = (const float*)d_in[3];
    const float* b_enc = (const float*)d_in[4];
    const float* w_m1  = (const float*)d_in[5];
    const float* b_m1  = (const float*)d_in[6];
    const float* w_m2  = (const float*)d_in[7];
    const float* b_m2  = (const float*)d_in[8];
    const float* w_i1  = (const float*)d_in[9];
    const float* b_i1  = (const float*)d_in[10];
    const float* w_i2  = (const float*)d_in[11];
    const float* b_i2  = (const float*)d_in[12];

    half_t* feat  = (half_t*)d_ws;
    half_t* w_m2T = (half_t*)d_ws + WM2T_OFF;
    half_t* w_i1T = (half_t*)d_ws + WI1T_OFF;

    const int conv_threads = BB * HH_ * WW_ * CC;            // 1,179,648
    liif_conv3x3<<<(conv_threads + 255) / 256, 256, 0, stream>>>(inp, w_enc, b_enc, feat);

    const int prep_threads = 1024 * 64 + 64 * 96;            // 71,680
    liif_prep_weights<<<(prep_threads + 255) / 256, 256, 0, stream>>>(w_m2, w_i1, w_m2T, w_i1T);

    const int nblocks = BB * (QQ / 16);                      // 4096
    liif_main<<<nblocks, 256, 0, stream>>>(inp, coord, cell, w_m1, b_m1, b_m2,
                                           b_i1, w_i2, b_i2, feat, w_m2T, w_i1T,
                                           (float*)d_out);
}
